// SelfAttention_44186623541813
// MI455X (gfx1250) — matrix operand order repaired and checked by reading, not compiled, not run on hardware
//
#include <hip/hip_runtime.h>
#include <hip/hip_bf16.h>

typedef __attribute__((ext_vector_type(16))) __bf16 v16bf;
typedef __attribute__((ext_vector_type(8)))  float  v8f;

#define E_DIM 1024
#define N_HEADS 16
#define SDIM 64
#define TSEQ 2048
#define BATCH 2
#define NROWS (BATCH * TSEQ) /* 4096 */

union Frag16 { v16bf v; unsigned u[8]; };

// A fragment (16x32 bf16) per ISA 7.12.2: lane l<16 -> row l, K 0..7 / 16..23;
// lane l>=16 -> row l-16, K 8..15 / 24..31.
__device__ inline v16bf ld_afrag(const __bf16* base, int stride, int row0, int k0) {
  const int lane = threadIdx.x & 31;
  const int r = lane & 15, hi = lane >> 4;
  const unsigned* p = (const unsigned*)(base + (size_t)(row0 + r) * stride + k0);
  Frag16 f;
#pragma unroll
  for (int j = 0; j < 4; ++j) { f.u[j] = p[hi * 4 + j]; f.u[4 + j] = p[8 + hi * 4 + j]; }
  return f.v;
}

// B fragment (32x16 bf16) from LDS stored [n][k]: lanes 0-15 K 0..7 / 16..23 col=lane,
// lanes 16-31 K 8..15 / 24..31 col=lane-16.
__device__ inline v16bf ld_bfrag(const __bf16* base, int stride, int col0, int k0) {
  const int lane = threadIdx.x & 31;
  const int c = lane & 15, hi = lane >> 4;
  const unsigned* p = (const unsigned*)(base + (size_t)(col0 + c) * stride + k0);
  Frag16 f;
#pragma unroll
  for (int j = 0; j < 4; ++j) { f.u[j] = p[hi * 4 + j]; f.u[4 + j] = p[8 + hi * 4 + j]; }
  return f.v;
}

__device__ inline v8f wmma_bf16(v16bf a, v16bf b, v8f c) {
  return __builtin_amdgcn_wmma_f32_16x16x32_bf16(false, a, false, b, (short)0, c,
                                                 false, false);
}

// ---- Tensor Data Mover: 2-D tile (rows x rowDw dwords) global -> LDS ----
// D# built per ISA 8.3/8.4: group0 {count=1, lds_addr, global_addr, type=2},
// group1 {data_size=4B, tensor_dim0=rowDw, tensor_dim1=rows, tile_dim0=rowDw,
//         tile_dim1=rows, tensor_dim0_stride=strideDw}. Groups 2/3 zero (2-D).
typedef unsigned tdm_u4 __attribute__((ext_vector_type(4)));
typedef int      tdm_i8 __attribute__((ext_vector_type(8)));
typedef int      tdm_i4 __attribute__((ext_vector_type(4)));

__device__ inline void tdm_load_tile(const void* gptr, unsigned lds_off,
                                     int rows, int rowDw, int strideDw) {
  const unsigned long long ga = (unsigned long long)gptr;
  tdm_u4 g0;
  g0[0] = 1u;                                   // count=1, user mode
  g0[1] = lds_off;                              // LDS byte address
  g0[2] = (unsigned)(ga & 0xffffffffu);         // global_addr[31:0]
  g0[3] = (unsigned)((ga >> 32) & 0x1ffffffu)   // global_addr[56:32]
          | (2u << 30);                         // type=2 ("image")
  tdm_i8 g1 = {};
  g1[0] = (int)(2u << 16);                                  // data_size=2 -> 4B units
  g1[1] = (int)(((unsigned)rowDw & 0xffffu) << 16);         // tensor_dim0 lo
  g1[2] = (int)((((unsigned)rowDw >> 16) & 0xffffu)         // tensor_dim0 hi
          | (((unsigned)rows & 0xffffu) << 16));            // tensor_dim1 lo
  g1[3] = (int)((((unsigned)rows >> 16) & 0xffffu)          // tensor_dim1 hi
          | (((unsigned)rowDw & 0xffffu) << 16));           // tile_dim0
  g1[4] = (int)((unsigned)rows & 0xffffu);                  // tile_dim1 (tile_dim2=0)
  g1[5] = (int)(unsigned)strideDw;                          // tensor_dim0_stride lo
  g1[6] = 0;                                                // stride hi / dim1_stride lo
  g1[7] = 0;
  tdm_i4 gz = {};
#if __clang_major__ >= 23
  tdm_i8 gz8 = {};
  __builtin_amdgcn_tensor_load_to_lds(g0, g1, gz, gz, gz8, 0);
#else
  __builtin_amdgcn_tensor_load_to_lds(g0, g1, gz, gz, 0);
#endif
}

__global__ __launch_bounds__(256) void cvt_kernel(const float* __restrict__ in,
                                                  __bf16* __restrict__ out, int n) {
  int i = blockIdx.x * 256 + threadIdx.x;
  if (i < n) out[i] = (__bf16)in[i];
}

// C(128x64 tile) = A(Nx1024 bf16) @ B(1024xNC bf16); per-mode epilogue:
//  0: per-64-col LayerNorm (head-dim kq-norm) -> bf16
//  1: plain -> bf16
//  2: +bias -> f32
__global__ __launch_bounds__(128) void gemm128_kernel(
    const __bf16* __restrict__ A, const __bf16* __restrict__ Bm,
    __bf16* __restrict__ outBf, float* __restrict__ outF,
    const float* __restrict__ lnw, const float* __restrict__ lnb,
    const float* __restrict__ bias, int Kdim, int NC, int mode) {
  __shared__ __align__(16) __bf16 As[128 * 32];  // [row][k]  (TDM-filled)
  __shared__ __align__(16) __bf16 Bs[64 * 32];   // [n][k]    (transposed stage)
  __shared__ __align__(16) float  Cs[128 * 64];

  const int tid = threadIdx.x;
  const int wave = tid >> 5;
  const int row0 = blockIdx.y * 128;
  const int col0 = blockIdx.x * 64;
  const unsigned ldsAs = (unsigned)(unsigned long long)(void*)As;

  v8f acc[2][4] = {};

  const int kp = tid & 15;            // k-pair: k = 2kp, 2kp+1
  const int bn0 = (tid >> 4) * 8;     // 8 n-values per thread
  unsigned* BsU = (unsigned*)Bs;

  for (int k0 = 0; k0 < Kdim; k0 += 32) {
    if (tid < 32)  // wave 0 drives the TDM for the A tile (128 rows x 64B)
      tdm_load_tile(A + (size_t)row0 * Kdim + k0, ldsAs, 128, 16, Kdim >> 1);
    {  // B tile transposed stage: pack two k-rows into dword stores
      union { uint4 q; __bf16 h[8]; } r0, r1;
      r0.q = *(const uint4*)(Bm + (size_t)(k0 + 2 * kp) * NC + col0 + bn0);
      r1.q = *(const uint4*)(Bm + (size_t)(k0 + 2 * kp + 1) * NC + col0 + bn0);
#pragma unroll
      for (int j = 0; j < 8; ++j) {
        union { __bf16 h[2]; unsigned u; } pk;
        pk.h[0] = r0.h[j]; pk.h[1] = r1.h[j];
        BsU[(bn0 + j) * 16 + kp] = pk.u;
      }
    }
    if (tid < 32) __builtin_amdgcn_s_wait_tensorcnt(0);
    __syncthreads();

    v16bf af0 = ld_afrag(As, 32, wave * 32, 0);
    v16bf af1 = ld_afrag(As, 32, wave * 32 + 16, 0);
#pragma unroll
    for (int f = 0; f < 4; ++f) {
      v16bf bfr = ld_bfrag(Bs, 32, f * 16, 0);
      acc[0][f] = wmma_bf16(af0, bfr, acc[0][f]);
      acc[1][f] = wmma_bf16(af1, bfr, acc[1][f]);
    }
    __syncthreads();
  }

  const int lane = tid & 31, li = lane & 15, hig = lane >> 4;
#pragma unroll
  for (int g = 0; g < 2; ++g)
#pragma unroll
    for (int f = 0; f < 4; ++f)
#pragma unroll
      for (int i = 0; i < 8; ++i)
        Cs[(wave * 32 + g * 16 + hig * 8 + i) * 64 + f * 16 + li] = acc[g][f][i];
  __syncthreads();

  // epilogue: one row per thread (128 rows, 128 threads)
  const float* rowp = Cs + tid * 64;
  const size_t gbase = (size_t)(row0 + tid) * NC + col0;
  if (mode == 0) {
    float mu = 0.f;
#pragma unroll
    for (int j = 0; j < 64; ++j) mu += rowp[j];
    mu *= (1.f / 64.f);
    float var = 0.f;
#pragma unroll
    for (int j = 0; j < 64; ++j) { float d = rowp[j] - mu; var += d * d; }
    var *= (1.f / 64.f);
    const float rs = rsqrtf(var + 1e-5f);
#pragma unroll
    for (int j = 0; j < 64; ++j)
      outBf[gbase + j] = (__bf16)((rowp[j] - mu) * rs * lnw[j] + lnb[j]);
  } else if (mode == 1) {
#pragma unroll
    for (int j = 0; j < 64; ++j) outBf[gbase + j] = (__bf16)rowp[j];
  } else {
#pragma unroll
    for (int j = 0; j < 64; ++j) outF[gbase + j] = rowp[j] + bias[col0 + j];
  }
}

// Flash-attention: one (batch, head, 64-q-row tile) per workgroup, 4 waves.
__global__ __launch_bounds__(128) void attn_kernel(
    const __bf16* __restrict__ Q, const __bf16* __restrict__ K,
    const __bf16* __restrict__ V, __bf16* __restrict__ O) {
  __shared__ __align__(16) __bf16 Qs[64 * 64];   // [q][s]    (TDM)
  __shared__ __align__(16) __bf16 Ks[64 * 64];   // [key][s]  (TDM)
  __shared__ __align__(16) __bf16 Vts[64 * 64];  // [s][key]  (manual transpose)
  __shared__ __align__(16) __bf16 Ps[64 * 64];   // [q][key]

  const int tid = threadIdx.x;
  const int wave = tid >> 5;
  const int lane = tid & 31, li = lane & 15, hig = lane >> 4;
  const int qt = blockIdx.x;
  const int h = blockIdx.y;
  const int b = blockIdx.z;
  const int qBase = qt * 64;
  const size_t bOff = (size_t)b * TSEQ;
  const unsigned ldsQs = (unsigned)(unsigned long long)(void*)Qs;
  const unsigned ldsKs = (unsigned)(unsigned long long)(void*)Ks;

  if (tid < 32) {
    tdm_load_tile(Q + (bOff + qBase) * E_DIM + h * SDIM, ldsQs, 64, 32, E_DIM / 2);
    __builtin_amdgcn_s_wait_tensorcnt(0);
  }

  v8f o[4] = {};
  float mrow[8], lrow[8];
#pragma unroll
  for (int i = 0; i < 8; ++i) { mrow[i] = -__builtin_inff(); lrow[i] = 0.f; }

  __syncthreads();
  const v16bf qa0 = ld_afrag(Qs, 64, wave * 16, 0);
  const v16bf qa1 = ld_afrag(Qs, 64, wave * 16, 32);

  const int kp = tid & 31;           // key pair: 2kp, 2kp+1
  const int sb0 = (tid >> 5) * 16;   // 16 s-values per thread
  unsigned* VtsU = (unsigned*)Vts;

  for (int kt = 0; kt <= qt; ++kt) {
    const int kBase = kt * 64;
    if (tid < 32)  // TDM stages the K tile while all waves transpose V
      tdm_load_tile(K + (bOff + kBase) * E_DIM + h * SDIM, ldsKs, 64, 32, E_DIM / 2);
    {  // V tile transposed stage: pack two key-rows, dword stores
      union { uint4 q[2]; __bf16 h[16]; } r0, r1;
      const uint4* v0 = (const uint4*)(V + (bOff + kBase + 2 * kp) * E_DIM + h * SDIM + sb0);
      const uint4* v1 = (const uint4*)(V + (bOff + kBase + 2 * kp + 1) * E_DIM + h * SDIM + sb0);
      r0.q[0] = v0[0]; r0.q[1] = v0[1];
      r1.q[0] = v1[0]; r1.q[1] = v1[1];
#pragma unroll
      for (int j = 0; j < 16; ++j) {
        union { __bf16 h[2]; unsigned u; } pk;
        pk.h[0] = r0.h[j]; pk.h[1] = r1.h[j];
        VtsU[(sb0 + j) * 32 + kp] = pk.u;
      }
    }
    if (tid < 32) __builtin_amdgcn_s_wait_tensorcnt(0);
    __syncthreads();

    // S = Q K^T  (K-dim = s = 64 -> 2 WMMA per fragment)
    v8f sf[4];
#pragma unroll
    for (int f = 0; f < 4; ++f) {
      v8f z = {};
      v16bf kb0 = ld_bfrag(Ks, 64, f * 16, 0);
      v16bf kb1 = ld_bfrag(Ks, 64, f * 16, 32);
      z = wmma_bf16(qa0, kb0, z);
      z = wmma_bf16(qa1, kb1, z);
      sf[f] = z;
    }

    // scale + causal mask (bites only on the diagonal tile)
    const int qr0 = qBase + wave * 16 + hig * 8;
#pragma unroll
    for (int f = 0; f < 4; ++f) {
      const int kc = kBase + f * 16 + li;
#pragma unroll
      for (int i = 0; i < 8; ++i) {
        float val = sf[f][i] * 0.125f;  // 1/sqrt(64)
        if (kc > qr0 + i) val = -__builtin_inff();
        sf[f][i] = val;
      }
    }

    // online softmax: per-row max over 4 frags, then 16-lane group reduce
    float rmax[8];
#pragma unroll
    for (int i = 0; i < 8; ++i)
      rmax[i] = fmaxf(fmaxf(sf[0][i], sf[1][i]), fmaxf(sf[2][i], sf[3][i]));
#pragma unroll
    for (int d = 1; d < 16; d <<= 1)
#pragma unroll
      for (int i = 0; i < 8; ++i)
        rmax[i] = fmaxf(rmax[i], __shfl_xor(rmax[i], d, 32));

    float mnew[8], alpha[8];
#pragma unroll
    for (int i = 0; i < 8; ++i) {
      mnew[i] = fmaxf(mrow[i], rmax[i]);
      alpha[i] = (mrow[i] == -__builtin_inff()) ? 0.f : __expf(mrow[i] - mnew[i]);
      mrow[i] = mnew[i];
    }

    float rsum[8];
#pragma unroll
    for (int i = 0; i < 8; ++i) rsum[i] = 0.f;
#pragma unroll
    for (int f = 0; f < 4; ++f)
#pragma unroll
      for (int i = 0; i < 8; ++i) {
        float p = __expf(sf[f][i] - mnew[i]);
        sf[f][i] = p;
        rsum[i] += p;
      }
#pragma unroll
    for (int d = 1; d < 16; d <<= 1)
#pragma unroll
      for (int i = 0; i < 8; ++i) rsum[i] += __shfl_xor(rsum[i], d, 32);
#pragma unroll
    for (int i = 0; i < 8; ++i) lrow[i] = lrow[i] * alpha[i] + rsum[i];

    // P (C-layout) -> LDS bf16 so it can be re-read in A-layout
#pragma unroll
    for (int f = 0; f < 4; ++f)
#pragma unroll
      for (int i = 0; i < 8; ++i)
        Ps[(wave * 16 + hig * 8 + i) * 64 + f * 16 + li] = (__bf16)sf[f][i];
    __syncthreads();

    // O = O*alpha + P @ V
#pragma unroll
    for (int f = 0; f < 4; ++f)
#pragma unroll
      for (int i = 0; i < 8; ++i) o[f][i] *= alpha[i];

    v16bf pa0 = ld_afrag(Ps, 64, wave * 16, 0);
    v16bf pa1 = ld_afrag(Ps, 64, wave * 16, 32);
#pragma unroll
    for (int f = 0; f < 4; ++f) {
      v16bf vb0 = ld_bfrag(Vts, 64, f * 16, 0);
      v16bf vb1 = ld_bfrag(Vts, 64, f * 16, 32);
      o[f] = wmma_bf16(pa0, vb0, o[f]);
      o[f] = wmma_bf16(pa1, vb1, o[f]);
    }
    __syncthreads();
  }

  // normalize and emit bf16 attention output (pre-Wo)
  float inv[8];
#pragma unroll
  for (int i = 0; i < 8; ++i) inv[i] = 1.f / lrow[i];
#pragma unroll
  for (int f = 0; f < 4; ++f)
#pragma unroll
    for (int i = 0; i < 8; ++i) {
      const size_t row = bOff + qBase + wave * 16 + hig * 8 + i;
      O[row * E_DIM + h * SDIM + f * 16 + li] = (__bf16)(o[f][i] * inv[i]);
    }
}

extern "C" void kernel_launch(void* const* d_in, const int* in_sizes, int n_in,
                              void* d_out, int out_size, void* d_ws, size_t ws_size,
                              hipStream_t stream) {
  (void)in_sizes; (void)n_in; (void)out_size; (void)ws_size;
  const float* x    = (const float*)d_in[0];
  const float* Wk   = (const float*)d_in[1];
  const float* Wq   = (const float*)d_in[2];
  const float* Wv   = (const float*)d_in[3];
  const float* Wo   = (const float*)d_in[4];
  const float* bo   = (const float*)d_in[5];
  const float* klnw = (const float*)d_in[6];
  const float* klnb = (const float*)d_in[7];
  const float* qlnw = (const float*)d_in[8];
  const float* qlnb = (const float*)d_in[9];
  float* out = (float*)d_out;

  char* ws = (char*)d_ws;
  const size_t NE = (size_t)NROWS * E_DIM;  // 4M elems
  const size_t EE = (size_t)E_DIM * E_DIM;  // 1M elems
  __bf16* xb  = (__bf16*)ws; ws += NE * 2;
  __bf16* Wkb = (__bf16*)ws; ws += EE * 2;
  __bf16* Wqb = (__bf16*)ws; ws += EE * 2;
  __bf16* Wvb = (__bf16*)ws; ws += EE * 2;
  __bf16* Wob = (__bf16*)ws; ws += EE * 2;
  __bf16* Qn  = (__bf16*)ws; ws += NE * 2;
  __bf16* Kn  = (__bf16*)ws; ws += NE * 2;
  __bf16* Vb  = (__bf16*)ws; ws += NE * 2;
  __bf16* Ob  = (__bf16*)ws; ws += NE * 2;  // total 48 MB

  cvt_kernel<<<(int)((NE + 255) / 256), 256, 0, stream>>>(x, xb, (int)NE);
  cvt_kernel<<<(int)((EE + 255) / 256), 256, 0, stream>>>(Wk, Wkb, (int)EE);
  cvt_kernel<<<(int)((EE + 255) / 256), 256, 0, stream>>>(Wq, Wqb, (int)EE);
  cvt_kernel<<<(int)((EE + 255) / 256), 256, 0, stream>>>(Wv, Wvb, (int)EE);
  cvt_kernel<<<(int)((EE + 255) / 256), 256, 0, stream>>>(Wo, Wob, (int)EE);

  dim3 gg(E_DIM / 64, NROWS / 128);
  gemm128_kernel<<<gg, 128, 0, stream>>>(xb, Wqb, Qn, nullptr, qlnw, qlnb, nullptr,
                                         E_DIM, E_DIM, 0);
  gemm128_kernel<<<gg, 128, 0, stream>>>(xb, Wkb, Kn, nullptr, klnw, klnb, nullptr,
                                         E_DIM, E_DIM, 0);
  gemm128_kernel<<<gg, 128, 0, stream>>>(xb, Wvb, Vb, nullptr, nullptr, nullptr,
                                         nullptr, E_DIM, E_DIM, 1);

  attn_kernel<<<dim3(TSEQ / 64, N_HEADS, BATCH), 128, 0, stream>>>(Qn, Kn, Vb, Ob);

  gemm128_kernel<<<gg, 128, 0, stream>>>(Ob, Wob, nullptr, out, nullptr, nullptr, bo,
                                         E_DIM, E_DIM, 2);
}
